// complexGRUCell_20650202759525
// MI455X (gfx1250) — matrix operand order repaired and checked by reading, not compiled, not run on hardware
//
#include <hip/hip_runtime.h>
#include <hip/hip_bf16.h>

// ---------------------------------------------------------------------------
// complex GRU cell on MI455X (gfx1250): f16 WMMA GEMMs + fused epilogues
// ---------------------------------------------------------------------------

typedef __attribute__((ext_vector_type(16))) _Float16 v16h;
typedef __attribute__((ext_vector_type(8)))  _Float16 v8h;
typedef __attribute__((ext_vector_type(8)))  float    v8f;

static constexpr int  kB  = 16384;
static constexpr int  kH  = 256;
static constexpr long kBH = (long)kB * kH;

// -------------------- weight / bias packing (phase-2 combined) --------------
// Wc1: [N=1024][K=1024] f16.  N blocks: r_re, r_im, z_re, z_im.
// K blocks: x_re, h_re, x_im, h_im.  U-terms folded into the h columns.
__global__ void pack_w1(const float* __restrict__ wr_re, const float* __restrict__ wr_im,
                        const float* __restrict__ wz_re, const float* __restrict__ wz_im,
                        const float* __restrict__ ur_re, const float* __restrict__ ur_im,
                        const float* __restrict__ uz_re, const float* __restrict__ uz_im,
                        const float* __restrict__ br_re, const float* __restrict__ br_im,
                        const float* __restrict__ bz_re, const float* __restrict__ bz_im,
                        _Float16* __restrict__ Wc, float* __restrict__ bias)
{
    int idx = blockIdx.x * blockDim.x + threadIdx.x;
    if (idx >= 1024 * 1024) return;
    int n = idx >> 10, k = idx & 1023;
    int j  = n & 255, nb = n >> 8;      // nb: 0 r_re, 1 r_im, 2 z_re, 3 z_im
    int kk = k & 255, kb = k >> 8;      // kb: 0 x_re, 1 h_re, 2 x_im, 3 h_im
    int gate = nb >> 1, imag = nb & 1;
    const float* Wre = gate ? wz_re : wr_re;   // [256][512]
    const float* Wim = gate ? wz_im : wr_im;
    const float* Ure = gate ? uz_re : ur_re;   // [256][256]
    const float* Uim = gate ? uz_im : ur_im;
    float w = 0.f;
    if (!imag) {
        if      (kb == 0) w =  Wre[j * 512 + kk];
        else if (kb == 1) w =  Wre[j * 512 + 256 + kk] + Ure[j * 256 + kk];
        else if (kb == 2) w = -Wim[j * 512 + kk];
        else              w = -(Wim[j * 512 + 256 + kk] + Uim[j * 256 + kk]);
    } else {
        if      (kb == 0) w =  Wim[j * 512 + kk];
        else if (kb == 1) w =  Wim[j * 512 + 256 + kk] + Uim[j * 256 + kk];
        else if (kb == 2) w =  Wre[j * 512 + kk];
        else              w =  Wre[j * 512 + 256 + kk] + Ure[j * 256 + kk];
    }
    Wc[idx] = (_Float16)w;
    if (k == 0) {
        const float* bre = gate ? bz_re : br_re;
        const float* bim = gate ? bz_im : br_im;
        bias[n] = imag ? (bre[j] + bim[j]) : (bre[j] - bim[j]);
    }
}

// Wc2: [N=512][K=1536] f16.  N blocks: e_re, e_im.
// K blocks: x_re, rh_re, x_im, rh_im, h_re, h_im.
__global__ void pack_w2(const float* __restrict__ wh_re, const float* __restrict__ wh_im,
                        const float* __restrict__ uh_re, const float* __restrict__ uh_im,
                        const float* __restrict__ bh_re, const float* __restrict__ bh_im,
                        _Float16* __restrict__ Wc, float* __restrict__ bias)
{
    int idx = blockIdx.x * blockDim.x + threadIdx.x;
    if (idx >= 512 * 1536) return;
    int n = idx / 1536, k = idx - n * 1536;
    int j = n & 255, imag = n >> 8;
    int kb = k >> 8, kk = k & 255;
    float w = 0.f;
    if (!imag) {
        if      (kb == 0) w =  wh_re[j * 512 + kk];
        else if (kb == 1) w =  wh_re[j * 512 + 256 + kk];
        else if (kb == 2) w = -wh_im[j * 512 + kk];
        else if (kb == 3) w = -wh_im[j * 512 + 256 + kk];
        else if (kb == 4) w =  uh_re[j * 256 + kk];
        else              w = -uh_im[j * 256 + kk];
    } else {
        if      (kb == 0) w =  wh_im[j * 512 + kk];
        else if (kb == 1) w =  wh_im[j * 512 + 256 + kk];
        else if (kb == 2) w =  wh_re[j * 512 + kk];
        else if (kb == 3) w =  wh_re[j * 512 + 256 + kk];
        else if (kb == 4) w =  uh_im[j * 256 + kk];
        else              w =  uh_re[j * 256 + kk];
    }
    Wc[idx] = (_Float16)w;
    if (k == 0)
        bias[n] = imag ? (bh_re[j] + bh_im[j]) : (bh_re[j] - bh_im[j]);
}

// -------------------- activation packing -----------------------------------
// V2: [B][1024] f16 = [x_re | h_re | x_im | h_im]
__global__ void pack_v2(const float* __restrict__ x_re, const float* __restrict__ x_im,
                        const float* __restrict__ h_re, const float* __restrict__ h_im,
                        _Float16* __restrict__ V2)
{
    long idx = (long)blockIdx.x * blockDim.x + threadIdx.x;
    if (idx >= (long)kB * 1024) return;
    int b = (int)(idx >> 10), k = (int)(idx & 1023);
    int kk = k & 255, kb = k >> 8;
    const float* src = (kb == 0) ? x_re : (kb == 1) ? h_re : (kb == 2) ? x_im : h_im;
    V2[idx] = (_Float16)src[(long)b * 256 + kk];
}

// After the r/z GEMM (sigmoid applied): compute rh = r*h (complex) and pack
// V3: [B][1536] f16 = [x_re | rh_re | x_im | rh_im | h_re | h_im]
__global__ void build_v3(const float* __restrict__ G2,
                         const float* __restrict__ x_re, const float* __restrict__ x_im,
                         const float* __restrict__ h_re, const float* __restrict__ h_im,
                         _Float16* __restrict__ V3)
{
    long idx = (long)blockIdx.x * blockDim.x + threadIdx.x;
    if (idx >= kBH) return;
    int b = (int)(idx >> 8), j = (int)(idx & 255);
    float rr = G2[(long)b * 1024 + j];
    float ri = G2[(long)b * 1024 + 256 + j];
    float hr = h_re[idx], hi = h_im[idx];
    float rhr = rr * hr - ri * hi;
    float rhi = rr * hi + ri * hr;
    long base = (long)b * 1536;
    V3[base +        j] = (_Float16)x_re[idx];
    V3[base + 256  + j] = (_Float16)rhr;
    V3[base + 512  + j] = (_Float16)x_im[idx];
    V3[base + 768  + j] = (_Float16)rhi;
    V3[base + 1024 + j] = (_Float16)hr;
    V3[base + 1280 + j] = (_Float16)hi;
}

// Final complex blend: h_new = (1-z)*h + z*h_hat
__global__ void finalize(const float* __restrict__ G2, const float* __restrict__ G3,
                         const float* __restrict__ h_re, const float* __restrict__ h_im,
                         float* __restrict__ out)
{
    long idx = (long)blockIdx.x * blockDim.x + threadIdx.x;
    if (idx >= kBH) return;
    int b = (int)(idx >> 8), j = (int)(idx & 255);
    float zr  = G2[(long)b * 1024 + 512 + j];
    float zi  = G2[(long)b * 1024 + 768 + j];
    float hhr = G3[(long)b * 512 + j];
    float hhi = G3[(long)b * 512 + 256 + j];
    float hr = h_re[idx], hi = h_im[idx];
    out[idx]       = (1.f - zr) * hr + zi * hi + zr * hhr - zi * hhi;
    out[kBH + idx] = (1.f - zr) * hi - zi * hr + zr * hhi + zi * hhr;
}

// -------------------- WMMA GEMM ---------------------------------------------
// C[M,N] = act( A[M,K] * W[N,K]^T + bias ),  A/W f16 row-major, C f32.
// Block = 256 threads (8 waves); 128x128 tile; wave = 32x64 (2x4 WMMA tiles).
__device__ __forceinline__ v16h load_a_frag(const _Float16* __restrict__ p, int ld, int lane)
{
    // ISA 16-bit A 16x32 layout: lanes 0-15 -> K 0..7 / 16..23, lanes 16-31 -> K 8..15 / 24..31
    int m  = lane & 15;
    int k0 = (lane & 16) ? 8 : 0;
    const _Float16* q = p + (long)m * ld + k0;
    v8h lo = *(const v8h*)(q);        // K = k0 .. k0+7
    v8h hi = *(const v8h*)(q + 16);   // K = k0+16 .. k0+23
    return __builtin_shufflevector(lo, hi, 0,1,2,3,4,5,6,7,8,9,10,11,12,13,14,15);
}

__device__ __forceinline__ v16h load_b_frag(const _Float16* __restrict__ p, int ld, int lane)
{
    // B 32x16 (KxN): lane = N, lanes 0-15 hold K 0..7 / 16..23, lanes 16-31 hold K 8..15 / 24..31
    int n  = lane & 15;
    int k0 = (lane & 16) ? 8 : 0;
    const _Float16* q = p + (long)n * ld + k0;
    v8h lo = *(const v8h*)(q);
    v8h hi = *(const v8h*)(q + 16);
    return __builtin_shufflevector(lo, hi, 0,1,2,3,4,5,6,7,8,9,10,11,12,13,14,15);
}

__global__ __launch_bounds__(256)
void gemm_wmma_f16(const _Float16* __restrict__ A, const _Float16* __restrict__ W,
                   const float* __restrict__ bias, float* __restrict__ C,
                   int M, int N, int K, int act)   // act: 0 none, 1 sigmoid, 2 tanh
{
    int lane = threadIdx.x & 31;
    int wid  = threadIdx.x >> 5;
    int wm = wid & 3;                 // 4 strips of 32 rows
    int wn = wid >> 2;                // 2 strips of 64 cols
    int m0 = blockIdx.y * 128 + wm * 32;
    int n0 = blockIdx.x * 128 + wn * 64;

    v8f acc[2][4];
    v8f zero = {};
#pragma unroll
    for (int i = 0; i < 2; ++i)
#pragma unroll
        for (int j = 0; j < 4; ++j) acc[i][j] = zero;

    for (int k = 0; k < K; k += 32) {
        v16h a0 = load_a_frag(A + (long)m0 * K + k,        K, lane);
        v16h a1 = load_a_frag(A + (long)(m0 + 16) * K + k, K, lane);
        v16h b0 = load_b_frag(W + (long)(n0)      * K + k, K, lane);
        v16h b1 = load_b_frag(W + (long)(n0 + 16) * K + k, K, lane);
        v16h b2 = load_b_frag(W + (long)(n0 + 32) * K + k, K, lane);
        v16h b3 = load_b_frag(W + (long)(n0 + 48) * K + k, K, lane);

        acc[0][0] = __builtin_amdgcn_wmma_f32_16x16x32_f16(false, a0, false, b0, (short)0, acc[0][0], false, false);
        acc[0][1] = __builtin_amdgcn_wmma_f32_16x16x32_f16(false, a0, false, b1, (short)0, acc[0][1], false, false);
        acc[0][2] = __builtin_amdgcn_wmma_f32_16x16x32_f16(false, a0, false, b2, (short)0, acc[0][2], false, false);
        acc[0][3] = __builtin_amdgcn_wmma_f32_16x16x32_f16(false, a0, false, b3, (short)0, acc[0][3], false, false);
        acc[1][0] = __builtin_amdgcn_wmma_f32_16x16x32_f16(false, a1, false, b0, (short)0, acc[1][0], false, false);
        acc[1][1] = __builtin_amdgcn_wmma_f32_16x16x32_f16(false, a1, false, b1, (short)0, acc[1][1], false, false);
        acc[1][2] = __builtin_amdgcn_wmma_f32_16x16x32_f16(false, a1, false, b2, (short)0, acc[1][2], false, false);
        acc[1][3] = __builtin_amdgcn_wmma_f32_16x16x32_f16(false, a1, false, b3, (short)0, acc[1][3], false, false);
    }

    // C/D layout: VGPR r -> M = r (lanes 0-15) or r+8 (lanes 16-31); N = lane&15
    int cm = (lane & 16) ? 8 : 0;
    int cn = lane & 15;
#pragma unroll
    for (int i = 0; i < 2; ++i)
#pragma unroll
        for (int j = 0; j < 4; ++j) {
            int col = n0 + j * 16 + cn;
            float bv = bias ? bias[col] : 0.f;
#pragma unroll
            for (int r = 0; r < 8; ++r) {
                float v = acc[i][j][r] + bv;
                if      (act == 1) v = 1.f / (1.f + __expf(-v));
                else if (act == 2) v = tanhf(v);
                C[(long)(m0 + i * 16 + cm + r) * N + col] = v;
            }
        }
}

// -------------------- host driver -------------------------------------------
extern "C" void kernel_launch(void* const* d_in, const int* in_sizes, int n_in,
                              void* d_out, int out_size, void* d_ws, size_t ws_size,
                              hipStream_t stream)
{
    const float* x_re = (const float*)d_in[0];
    const float* x_im = (const float*)d_in[1];
    const float* h_re = (const float*)d_in[2];
    const float* h_im = (const float*)d_in[3];
    const float* wr_w_re = (const float*)d_in[4];
    const float* wr_w_im = (const float*)d_in[5];
    const float* wr_b_re = (const float*)d_in[6];
    const float* wr_b_im = (const float*)d_in[7];
    const float* wz_w_re = (const float*)d_in[8];
    const float* wz_w_im = (const float*)d_in[9];
    const float* wz_b_re = (const float*)d_in[10];
    const float* wz_b_im = (const float*)d_in[11];
    const float* wh_w_re = (const float*)d_in[12];
    const float* wh_w_im = (const float*)d_in[13];
    const float* wh_b_re = (const float*)d_in[14];
    const float* wh_b_im = (const float*)d_in[15];
    const float* ur_w_re = (const float*)d_in[16];
    const float* ur_w_im = (const float*)d_in[17];
    const float* uz_w_re = (const float*)d_in[18];
    const float* uz_w_im = (const float*)d_in[19];
    const float* uh_w_re = (const float*)d_in[20];
    const float* uh_w_im = (const float*)d_in[21];

    char* ws = (char*)d_ws;
    size_t off = 0;
    auto alloc = [&](size_t bytes) -> void* {
        void* p = ws + off;
        off = (off + bytes + 255) & ~(size_t)255;
        return p;
    };
    _Float16* Wc1  = (_Float16*)alloc((size_t)1024 * 1024 * sizeof(_Float16));
    float*    bias1= (float*)   alloc((size_t)1024 * sizeof(float));
    _Float16* Wc2  = (_Float16*)alloc((size_t)512 * 1536 * sizeof(_Float16));
    float*    bias2= (float*)   alloc((size_t)512 * sizeof(float));
    _Float16* V2   = (_Float16*)alloc((size_t)kB * 1024 * sizeof(_Float16));
    float*    G2   = (float*)   alloc((size_t)kB * 1024 * sizeof(float));
    _Float16* V3   = (_Float16*)alloc((size_t)kB * 1536 * sizeof(_Float16));
    float*    G3   = (float*)   alloc((size_t)kB * 512 * sizeof(float));
    (void)ws_size; (void)in_sizes; (void)n_in; (void)out_size;

    // 1. pack combined weights / biases (tiny)
    pack_w1<<<(1024 * 1024 + 255) / 256, 256, 0, stream>>>(
        wr_w_re, wr_w_im, wz_w_re, wz_w_im, ur_w_re, ur_w_im, uz_w_re, uz_w_im,
        wr_b_re, wr_b_im, wz_b_re, wz_b_im, Wc1, bias1);
    pack_w2<<<(512 * 1536 + 255) / 256, 256, 0, stream>>>(
        wh_w_re, wh_w_im, uh_w_re, uh_w_im, wh_b_re, wh_b_im, Wc2, bias2);

    // 2. pack V2 (f32 -> f16)
    pack_v2<<<(int)(((long)kB * 1024 + 255) / 256), 256, 0, stream>>>(x_re, x_im, h_re, h_im, V2);

    // 3. r/z gates: [16384,1024] x [1024,1024]^T, sigmoid epilogue
    gemm_wmma_f16<<<dim3(1024 / 128, kB / 128), 256, 0, stream>>>(
        V2, Wc1, bias1, G2, kB, 1024, 1024, 1);

    // 4. rh = r*h (complex), build V3
    build_v3<<<(int)((kBH + 255) / 256), 256, 0, stream>>>(G2, x_re, x_im, h_re, h_im, V3);

    // 5. candidate gate: [16384,1536] x [1536,512]^T, tanh epilogue
    gemm_wmma_f16<<<dim3(512 / 128, kB / 128), 256, 0, stream>>>(
        V3, Wc2, bias2, G3, kB, 512, 1536, 2);

    // 6. h_new = (1-z)*h + z*h_hat (complex)
    finalize<<<(int)((kBH + 255) / 256), 256, 0, stream>>>(G2, G3, h_re, h_im, (float*)d_out);
}
